// Elman_model_4380866641998
// MI455X (gfx1250) — matrix operand order repaired and checked by reading, not compiled, not run on hardware
//
#include <hip/hip_runtime.h>
#include <hip/hip_bf16.h>

typedef __attribute__((ext_vector_type(16))) __bf16 v16bf;
typedef __attribute__((ext_vector_type(8)))  float  v8f;

#define T_STEPS 4096
#define BATCH   64
#define IN_DIM  4
#define HID     256
#define OUT_DIM 2
#define KCH     9            // 9 K-chunks of 32: 256 hidden + (4 input + 1 bias + pad)
#define AROW    296          // padded LDS row stride in bf16 elems (148 dwords -> conflict-free)
#define NWAVE   17           // 16 hidden-column strips + 1 head wave (Wf columns)
#define NTHR    (NWAVE * 32)

// Load one 16x32 bf16 A-fragment row slice from LDS per the CDNA5 A layout:
// lanes 0-15: V0-3 = K k0..k0+7, V4-7 = K k0+16..k0+23 ; lanes 16-31 use k0+8 base.
__device__ __forceinline__ v16bf lds_load_afrag(const __bf16* rowp, int k0) {
    union { uint4 u[2]; v16bf v; } U;
    U.u[0] = *(const uint4*)(rowp + k0);
    U.u[1] = *(const uint4*)(rowp + k0 + 16);
    return U.v;
}

// One augmented GEMM pass: acc[b, n] = sum_{k<288} A[b,k] * B[k,n]
// 3-term bf16 hi/lo split for near-fp32 accuracy; A loads software-pipelined.
__device__ __forceinline__ v8f step_gemm(const __bf16* __restrict__ arow_hi,
                                         const __bf16* __restrict__ arow_lo,
                                         const v16bf* Bh, const v16bf* Bl,
                                         int lhalf) {
    v8f acc = {};
    v16bf ah = lds_load_afrag(arow_hi, 8 * lhalf);
    v16bf al = lds_load_afrag(arow_lo, 8 * lhalf);
    #pragma unroll
    for (int c = 0; c < KCH; ++c) {
        v16bf ah_n = ah, al_n = al;
        if (c + 1 < KCH) {                       // prefetch next chunk under this chunk's WMMAs
            const int k0 = 32 * (c + 1) + 8 * lhalf;
            ah_n = lds_load_afrag(arow_hi, k0);
            al_n = lds_load_afrag(arow_lo, k0);
        }
        acc = __builtin_amdgcn_wmma_f32_16x16x32_bf16(false, ah, false, Bh[c], (short)0, acc, false, false);
        acc = __builtin_amdgcn_wmma_f32_16x16x32_bf16(false, ah, false, Bl[c], (short)0, acc, false, false);
        acc = __builtin_amdgcn_wmma_f32_16x16x32_bf16(false, al, false, Bh[c], (short)0, acc, false, false);
        ah = ah_n; al = al_n;
    }
    return acc;
}

__global__ __launch_bounds__(NTHR, 1)
void elman_scan_kernel(const float* __restrict__ input,
                       const float* __restrict__ Wi,
                       const float* __restrict__ bi,
                       const float* __restrict__ Wh,
                       const float* __restrict__ bhid,
                       const float* __restrict__ Wf,
                       const float* __restrict__ bfin,
                       float* __restrict__ out) {
    // double-buffered augmented state: cols 0..255 = h, 256..259 = x_t, 260 = 1.0
    __shared__ __bf16 sHhi[2][16 * AROW];
    __shared__ __bf16 sHlo[2][16 * AROW];

    const int tid   = threadIdx.x;
    const int wv    = tid >> 5;               // waves 0..15: hidden strips; wave 16: head
    const int lane  = tid & 31;
    const int lmod  = lane & 15;
    const int lhalf = lane >> 4;
    const int b0    = blockIdx.x * 16;        // batch rows of this workgroup
    const int j     = wv * 16 + lmod;         // hidden column (waves 0..15 only)

    // ---- one-time: build B operand in registers, split bf16 hi/lo (144 VGPRs) ----
    // waves 0..15: columns of [Wh | Wi | bi+bh] ; wave 16: columns of [Wf | 0 | bf]
    v16bf Bh[KCH], Bl[KCH];
    #pragma unroll
    for (int c = 0; c < KCH; ++c) {
        v16bf vh, vl;
        #pragma unroll
        for (int e = 0; e < 16; ++e) {
            const int k = 32 * c + (e < 8 ? 8 * lhalf + e : 16 + 8 * lhalf + (e - 8));  // B layout: lanes<16 K=0..7,16..23; lanes>=16 K=8..15,24..31
            float v = 0.0f;
            if (wv < 16) {
                if (k < HID)                  v = Wh[(size_t)j * HID + k];     // out[b,j]=sum_k h[b,k]*Wh[j,k]
                else if (k < HID + IN_DIM)    v = Wi[j * IN_DIM + (k - HID)];  // input projection
                else if (k == HID + IN_DIM)   v = bi[j] + bhid[j];             // fused bias (vs const-1 col)
            } else if (lmod < OUT_DIM) {
                if (k < HID)                  v = Wf[lmod * HID + k];          // head: out[b,o]=sum_k h[b,k]*Wf[o,k]
                else if (k == HID + IN_DIM)   v = bfin[lmod];                  // head bias
            }
            const __bf16 hi = (__bf16)v;
            vh[e] = hi;
            vl[e] = (__bf16)(v - (float)hi);
        }
        Bh[c] = vh; Bl[c] = vl;
    }

    // ---- init: h_{-1} = 0 in both buffers, constant-1 column, x_0 into buffer 0 ----
    for (int i = tid; i < 16 * AROW; i += NTHR) {
        sHhi[0][i] = (__bf16)0.0f;  sHhi[1][i] = (__bf16)0.0f;
        sHlo[0][i] = (__bf16)0.0f;  sHlo[1][i] = (__bf16)0.0f;
    }
    __syncthreads();
    if (tid < 16) {
        sHhi[0][tid * AROW + (HID + IN_DIM)] = (__bf16)1.0f;
        sHhi[1][tid * AROW + (HID + IN_DIM)] = (__bf16)1.0f;
        const float* xp = input + ((size_t)0 * BATCH + (b0 + tid)) * IN_DIM;
        #pragma unroll
        for (int i = 0; i < IN_DIM; ++i) {
            const float v = xp[i];
            const __bf16 hi = (__bf16)v;
            sHhi[0][tid * AROW + HID + i] = hi;
            sHlo[0][tid * AROW + HID + i] = (__bf16)(v - (float)hi);
        }
    }
    __syncthreads();

    for (int t = 0; t < T_STEPS; ++t) {
        const int rb = t & 1, wb = rb ^ 1;
        const __bf16* arow_hi = &sHhi[rb][lmod * AROW];   // A layout: lane carries row M = lane%16
        const __bf16* arow_lo = &sHlo[rb][lmod * AROW];

        const v8f acc = step_gemm(arow_hi, arow_lo, Bh, Bl, lhalf);

        if (wv == 16) {
            // head wave: acc = Wf*h_{t-1} + bf  ->  out[t-1]  (one step behind by construction)
            if (t > 0 && lmod < OUT_DIM) {
                #pragma unroll
                for (int r = 0; r < 8; ++r) {
                    const int b = b0 + r + 8 * lhalf;     // C/D layout: row = r + 8*(lane/16)
                    out[(((size_t)(t - 1) * BATCH) + b) * OUT_DIM + lmod] = tanhf(acc[r]);
                }
            }
            // stage x_{t+1} into the write buffer's augmented columns
            if (lane < 16 && t + 1 < T_STEPS) {
                const float* xp = input + ((size_t)(t + 1) * BATCH + (b0 + lane)) * IN_DIM;
                #pragma unroll
                for (int i = 0; i < IN_DIM; ++i) {
                    const float v = xp[i];
                    const __bf16 hi = (__bf16)v;
                    sHhi[wb][lane * AROW + HID + i] = hi;
                    sHlo[wb][lane * AROW + HID + i] = (__bf16)(v - (float)hi);
                }
            }
            if (lane == 0 && t + 2 < T_STEPS) {
                __builtin_prefetch(input + ((size_t)(t + 2) * BATCH + b0) * IN_DIM, 0, 1);
            }
        } else {
            // hidden waves: h_t = tanh(acc), publish split bf16 into the write buffer
            #pragma unroll
            for (int r = 0; r < 8; ++r) {
                const float hv = tanhf(acc[r]);
                const int bl = r + 8 * lhalf;
                const __bf16 hi = (__bf16)hv;
                sHhi[wb][bl * AROW + j] = hi;
                sHlo[wb][bl * AROW + j] = (__bf16)(hv - (float)hi);
            }
        }

        __syncthreads();   // single barrier: h_t / x_{t+1} visible, everyone done reading rb
    }

    // ---- final head pass: out[T-1] from h_{T-1} (sits in buffer T_STEPS&1 = 0) ----
    if (wv == 16) {
        const int rb = T_STEPS & 1;
        const v8f acc = step_gemm(&sHhi[rb][lmod * AROW], &sHlo[rb][lmod * AROW], Bh, Bl, lhalf);
        if (lmod < OUT_DIM) {
            #pragma unroll
            for (int r = 0; r < 8; ++r) {
                const int b = b0 + r + 8 * lhalf;
                out[(((size_t)(T_STEPS - 1) * BATCH) + b) * OUT_DIM + lmod] = tanhf(acc[r]);
            }
        }
    }
}

extern "C" void kernel_launch(void* const* d_in, const int* in_sizes, int n_in,
                              void* d_out, int out_size, void* d_ws, size_t ws_size,
                              hipStream_t stream) {
    const float* input = (const float*)d_in[0];
    // d_in[1] = target: unused by the forward reference
    const float* Wi   = (const float*)d_in[2];
    const float* bi   = (const float*)d_in[3];
    const float* Wh   = (const float*)d_in[4];
    const float* bh   = (const float*)d_in[5];
    const float* Wf   = (const float*)d_in[6];
    const float* bf   = (const float*)d_in[7];
    float* out = (float*)d_out;

    // 4 independent workgroups: batch rows split 16 each (recurrence couples only H)
    elman_scan_kernel<<<dim3(4), dim3(NTHR), 0, stream>>>(input, Wi, bi, Wh, bh, Wf, bf, out);
}
